// LocalCorrelation_7687991460603
// MI455X (gfx1250) — matrix operand order repaired and checked by reading, not compiled, not run on hardware
//
#include <hip/hip_runtime.h>
#include <hip/hip_bf16.h>

typedef __attribute__((ext_vector_type(16))) __bf16 v16bf;
typedef __attribute__((ext_vector_type(8)))  float  v8f;

#define B_  8
#define C_  256
#define H_  128
#define W_  192
#define R_  4
#define ND_ 81              // (2R+1)^2 displacements
#define HP_ (H_ + 2 * R_)   // 136: f2 padded height (+-R rows)
#define WP_ 208             // f2 padded width: 4 left + 12 right pad
                            // (N-tiles reach padded x = x0+79 <= 207)

// ---------------------------------------------------------------------------
// Zero-fill kernel (for the padded f2 buffer halo; zeroing everything is
// simplest and deterministic — interior is overwritten by the norm pass).
// ---------------------------------------------------------------------------
__global__ __launch_bounds__(256) void zero16_kernel(uint4* __restrict__ p,
                                                     size_t n16) {
  const size_t i = (size_t)blockIdx.x * blockDim.x + threadIdx.x;
  if (i < n16) p[i] = make_uint4(0u, 0u, 0u, 0u);
}

// ---------------------------------------------------------------------------
// Pass 1: L2-normalize over C, transpose CHW -> HWC, convert fp32 -> bf16,
// writing into a (possibly padded) HWC layout with row pitch Wp and offset
// `pad` in both y and x. One block: 32 consecutive x pixels at fixed (b, y).
// LDS tile stride 257 floats -> 257 % 64 == 1, bank-conflict free.
// ---------------------------------------------------------------------------
__global__ __launch_bounds__(256) void norm_hwc_bf16_kernel(
    const float* __restrict__ in, __bf16* __restrict__ out, int Hp, int Wp,
    int pad) {
  __shared__ float tile[32 * 257];
  __shared__ float partial[8][32];
  __shared__ float invn[32];

  const int t  = threadIdx.x;
  const int x0 = blockIdx.x * 32;
  const int y  = blockIdx.y;
  const int b  = blockIdx.z;

  const int xl = t & 31;   // pixel within tile
  const int cg = t >> 5;   // channel group (wave id)

  // Coalesced strided gather: each wave reads one channel row of 32 x.
  const float* src = in + (size_t)b * C_ * H_ * W_ + (size_t)y * W_ + x0;
  for (int c0 = 0; c0 < C_; c0 += 8) {
    const int c = c0 + cg;
    tile[xl * 257 + c] = src[(size_t)c * (H_ * W_) + xl];
  }
  __syncthreads();

  // Per-pixel sum of squares: 8 partial sums per pixel, tree-combined.
  float s = 0.0f;
  for (int c = cg * 32; c < cg * 32 + 32; ++c) {
    const float v = tile[xl * 257 + c];
    s += v * v;
  }
  partial[cg][xl] = s;
  __syncthreads();
  if (t < 32) {
    float acc = 0.0f;
#pragma unroll
    for (int p = 0; p < 8; ++p) acc += partial[p][t];
    invn[t] = 1.0f / fmaxf(sqrtf(acc), 1e-12f);
  }
  __syncthreads();

  // Write HWC bf16: 256 threads cover the 256 channels of one pixel per iter.
  __bf16* dst = out + ((size_t)(b * Hp + y + pad) * Wp + (x0 + pad)) * (size_t)C_;
  for (int xi = 0; xi < 32; ++xi) {
    dst[(size_t)xi * C_ + t] = (__bf16)(tile[xi * 257 + t] * invn[xi]);
  }
}

// ---------------------------------------------------------------------------
// Pass 2: correlation volume via v_wmma_f32_16x16x32_bf16.
// Block (256 thr = 8 waves) owns a 64-pixel x-strip at (b, y).
// Wave w -> (M-tile m = w/2, N-offset noff = w&1).
//   D[M][N] = <f1[x0+16m+M], f2[x0-4+16(m+noff)+N]>  (dot over C=256)
//   dx = 16*noff + N - M - 4; band |dx|<=4 covers all 9 dx, each exactly once.
// f2 is pre-padded with zeros, so all loads are unconditional. A sched
// barrier keeps all 16 B-loads ahead of the 8 WMMAs so the compiler emits
// staged partial s_wait_loadcnt instead of full waits per step.
// ---------------------------------------------------------------------------
__global__ __launch_bounds__(256) void corr_wmma_kernel(
    const __bf16* __restrict__ f1, const __bf16* __restrict__ f2p,
    float* __restrict__ out) {
  const int t    = threadIdx.x;
  const int lane = t & 31;
  const int w    = t >> 5;
  const int m    = w >> 1;
  const int noff = w & 1;

  const int x0 = blockIdx.x * 64;
  const int y  = blockIdx.y;
  const int b  = blockIdx.z;

  const int hl   = lane >> 4;   // lane half (0: lanes 0-15, 1: lanes 16-31)
  const int lmod = lane & 15;

  // ---- A fragments (f1, dy-invariant): ISA 16-bit A layout:
  //  lanes 0-15 : M=lane,    K = {0..7, 16..23}  (+32 per step)
  //  lanes 16-31: M=lane-16, K = {8..15, 24..31}
  const __bf16* f1p =
      f1 + ((size_t)((b * H_ + y) * W_) + (size_t)(x0 + 16 * m + lmod)) * C_;
  v16bf afrag[8];
#pragma unroll
  for (int s = 0; s < 8; ++s) {
    const int k0 = s * 32 + hl * 8;
    ((uint4*)&afrag[s])[0] = *(const uint4*)(f1p + k0);        // K k0..k0+7
    ((uint4*)&afrag[s])[1] = *(const uint4*)(f1p + k0 + 16);   // K k0+16..+23
  }

  // B column this lane owns, in padded coordinates (always in-bounds):
  // padded x = (x0 - R + 16*(m+noff) + lmod) + R
  const int f2xp = x0 + 16 * (m + noff) + lmod;
  const __bf16* f2base =
      f2p + ((size_t)(b * HP_) * WP_ + (size_t)f2xp) * C_;

  for (int dyi = 0; dyi < 2 * R_ + 1; ++dyi) {
    // padded row = (y + dy) + R = y + dyi, always in [0, HP_)
    const __bf16* f2pix = f2base + (size_t)(y + dyi) * WP_ * C_;
    __builtin_prefetch(f2pix + (size_t)WP_ * C_, 0, 1);  // next dy row

    // Load all 8 B fragments (one big load clause)...
    v16bf bfrag[8];
#pragma unroll
    for (int s = 0; s < 8; ++s) {
      // 16-bit B layout (32x16), matching A: lanes 0-15 hold K {0..7, 16..23},
      // lanes 16-31 hold K {8..15, 24..31} for column N = lane%16.
      const int k0 = s * 32 + hl * 8;
      ((uint4*)&bfrag[s])[0] = *(const uint4*)(f2pix + k0);       // K k0..+7
      ((uint4*)&bfrag[s])[1] = *(const uint4*)(f2pix + k0 + 16);  // K k0+16..+23
    }

    // ...then run the WMMA chain; nothing may cross this point, so loads
    // stay in flight and each WMMA gets a staged partial wait.
    __builtin_amdgcn_sched_barrier(0);

    v8f acc = {};
#pragma unroll
    for (int s = 0; s < 8; ++s) {
      acc = __builtin_amdgcn_wmma_f32_16x16x32_bf16(
          /*neg_a=*/false, afrag[s], /*neg_b=*/false, bfrag[s],
          /*c_mod=*/(short)0, acc, /*reuse_a=*/false, /*reuse_b=*/false);
    }

    // Scatter the valid diagonal band. D layout: lane l, reg r ->
    // M = r + 8*(l>=16), N = l%16.
#pragma unroll
    for (int r = 0; r < 8; ++r) {
      const int M  = r + hl * 8;
      const int dx = 16 * noff + lmod - M - R_;
      if ((unsigned)(dx + R_) <= (unsigned)(2 * R_)) {
        const int x = x0 + 16 * m + M;
        const int d = dyi * (2 * R_ + 1) + (dx + R_);
        out[(((size_t)(b * ND_ + d)) * H_ + y) * W_ + x] = acc[r];
      }
    }
  }
}

extern "C" void kernel_launch(void* const* d_in, const int* in_sizes, int n_in,
                              void* d_out, int out_size, void* d_ws,
                              size_t ws_size, hipStream_t stream) {
  const float* feat1 = (const float*)d_in[0];
  const float* feat2 = (const float*)d_in[1];

  // Workspace layout:
  //   f1n : B*H*W*C bf16   (unpadded HWC)      = 100.7 MB
  //   f2n : B*HP*WP*C bf16 (zero-padded HWC)   = 115.9 MB
  __bf16* f1n = (__bf16*)d_ws;
  __bf16* f2n = f1n + (size_t)B_ * H_ * W_ * C_;

  dim3 blk(256);

  // Zero the padded f2 buffer (halo must be zero; runs every call so the
  // result is independent of prior buffer state).
  const size_t f2_bytes = (size_t)B_ * HP_ * WP_ * C_ * sizeof(__bf16);
  const size_t n16      = f2_bytes / 16;
  zero16_kernel<<<(unsigned)((n16 + 255) / 256), blk, 0, stream>>>(
      (uint4*)f2n, n16);

  dim3 g1(W_ / 32, H_, B_);
  norm_hwc_bf16_kernel<<<g1, blk, 0, stream>>>(feat1, f1n, H_, W_, 0);
  norm_hwc_bf16_kernel<<<g1, blk, 0, stream>>>(feat2, f2n, HP_, WP_, R_);

  dim3 g2(W_ / 64, H_, B_);
  corr_wmma_kernel<<<g2, blk, 0, stream>>>(f1n, f2n, (float*)d_out);
}
